// AdaptiveFeaturePropagation_10316511445548
// MI455X (gfx1250) — matrix operand order repaired and checked by reading, not compiled, not run on hardware
//
#include <hip/hip_runtime.h>
#include <hip/hip_bf16.h>
#include <cstdint>
#include <cstddef>

typedef __attribute__((ext_vector_type(16))) _Float16 v16h;
typedef __attribute__((ext_vector_type(8)))  _Float16 v8h;
typedef __attribute__((ext_vector_type(8)))  float    v8f;

#define H 129
#define W 129
#define NPIX (H * W)       // 16641
#define PW 131             // padded width/height (1-pixel zero halo)
#define PPIX (PW * PW)     // 17161
#define XT 9               // ceil(129/16) x-tiles per row

// ---------------------------------------------------------------------------
// Bilinear upsample 33x33 -> 129x129 (half-pixel / align_corners=False).
// NCHW f32 in  ->  zero-padded HWC f16 out: [b][131][131][128].
// Threads map over PADDED coords; halo pixels get exact zeros, so the WMMA
// conv below needs no bounds checks.
// ---------------------------------------------------------------------------
__global__ void upsample_k(const float* __restrict__ in, _Float16* __restrict__ out, int total) {
    int idx = blockIdx.x * blockDim.x + threadIdx.x;
    if (idx >= total) return;
    int c = idx & 127;
    int t = idx >> 7;
    int px = t % PW; t /= PW;
    int py = t % PW;
    int b  = t / PW;
    _Float16* op = out + ((size_t)b * PPIX + (size_t)py * PW + px) * 128 + c;
    if (px == 0 || px == PW - 1 || py == 0 || py == PW - 1) { *op = (_Float16)0.0f; return; }
    int x = px - 1, y = py - 1;
    const float scale = 33.0f / 129.0f;
    float fx = (x + 0.5f) * scale - 0.5f;
    float fy = (y + 0.5f) * scale - 0.5f;
    int x0 = (int)floorf(fx), y0 = (int)floorf(fy);
    float wx = fx - (float)x0, wy = fy - (float)y0;
    int x0c = min(max(x0, 0), 32), x1c = min(max(x0 + 1, 0), 32);
    int y0c = min(max(y0, 0), 32), y1c = min(max(y0 + 1, 0), 32);
    const float* p = in + ((size_t)(b * 128 + c)) * (33 * 33);
    float v00 = p[y0c * 33 + x0c], v01 = p[y0c * 33 + x1c];
    float v10 = p[y1c * 33 + x0c], v11 = p[y1c * 33 + x1c];
    float v = v00 * (1.f - wy) * (1.f - wx) + v01 * (1.f - wy) * wx +
              v10 * wy * (1.f - wx)        + v11 * wy * wx;
    *op = (_Float16)v;
}

// ---------------------------------------------------------------------------
// Vectorized zero fill (16B per thread) — clears the padded concat buffer so
// its halo is zero before conv_reduce writes the interior.
// ---------------------------------------------------------------------------
__global__ void zero_fill_k(uint4* __restrict__ p, int totalVec) {
    int i = blockIdx.x * blockDim.x + threadIdx.x;
    if (i < totalVec) p[i] = make_uint4(0u, 0u, 0u, 0u);
}

// ---------------------------------------------------------------------------
// Repack OIHW f32 weights -> [cout][tap(ky*3+kx)][cin] f16 for WMMA A-frags.
// ---------------------------------------------------------------------------
__global__ void pack_w_k(const float* __restrict__ w, _Float16* __restrict__ o, int O, int I) {
    int idx = blockIdx.x * blockDim.x + threadIdx.x;
    int total = O * I * 9;
    if (idx >= total) return;
    int cin  = idx % I;
    int r    = (idx / I) % 9;
    int cout = idx / (I * 9);
    int ky = r / 3, kx = r % 3;
    o[idx] = (_Float16)w[(((size_t)cout * I + cin) * 3 + ky) * 3 + kx];
}

// ---------------------------------------------------------------------------
// Implicit-GEMM 3x3 conv (pad 1) via v_wmma_f32_16x16x32_f16, branch-free.
// Input  : zero-padded HWC f16 [img][PW][PW][CIN]
// Weights: f16 [cout][9][CIN]
// Output : HWC f16, padded (OUTPAD=1, stride PW, +1 halo offset) or tight
//          (OUTPAD=0, stride W); channel offset chOff; ReLU fused.
// Each wave: 16(cout) x 16(pixels) tile, K = 9 taps x CIN in chunks of 32.
// Fragment layouts used here (wave32; A/B use contiguous K, not the ISA table order):
//   A: lane M = lane&15; halves 0..15 -> K 0..15 (lanes<16) or 16..31 (hi)
//   B: lane N = lane&15; halves 0..15 -> K 0..15 (lanes<16) or 16..31 (hi)
//   C/D: elem i -> row i+(hi?8:0), col lane&15
// Lanes whose pixel x >= 129 (last x-tile) read a clamped in-bounds column;
// their D columns are never stored, so the garbage is inert.
// ---------------------------------------------------------------------------
template <int CIN, int COUTSTRIDE, int OUTPAD>
__global__ __launch_bounds__(256)
void conv3x3_wmma_k(const _Float16* __restrict__ in, const _Float16* __restrict__ wts,
                    const float* __restrict__ bias, _Float16* __restrict__ out, int chOff) {
    const int lane = threadIdx.x & 31;
    const int wv   = threadIdx.x >> 5;
    const int hi   = (lane >= 16) ? 1 : 0;
    const int n    = lane & 15;

    const int tile = blockIdx.x;                  // 0 .. H*XT-1
    const int y    = tile / XT;                   // output row (unpadded coords)
    const int x0   = (tile % XT) * 16;
    const int coutBase = (blockIdx.y * 8 + wv) * 16;
    const int img  = blockIdx.z;

    in  += (size_t)img * PPIX * CIN;
    out += (size_t)img * (OUTPAD ? (size_t)PPIX : (size_t)NPIX) * COUTSTRIDE + chOff;

    v8f acc;
#pragma unroll
    for (int i = 0; i < 8; ++i) acc[i] = bias[coutBase + i + (hi ? 8 : 0)];

    const int xb = x0 + n;                        // this lane's output column
    const int xs = xb < (W - 1) ? xb : (W - 1);   // clamped read column (in-bounds)
    const _Float16* wrow = wts + (size_t)(coutBase + n) * (9 * CIN) + (hi ? 16 : 0);
    // padded input coord of tap (ky,kx) for output (y,x): (y+ky, x+kx)
    const _Float16* brow = in + ((size_t)y * PW + xs) * CIN + (hi ? 16 : 0);

    for (int r = 0; r < 9; ++r) {
        const int ky = r / 3, kx = r % 3;
        const _Float16* bp = brow + ((size_t)ky * PW + kx) * CIN;
        const _Float16* ap = wrow + r * CIN;
        if (r < 8) __builtin_prefetch(ap + CIN, 0, 3);   // next tap's weights
#pragma unroll
        for (int c0 = 0; c0 < CIN; c0 += 32) {
            v16h a;
            v8h alo = *(const v8h*)(ap + c0);
            v8h ahi = *(const v8h*)(ap + c0 + 8);
#pragma unroll
            for (int e = 0; e < 8; ++e) { a[e] = alo[e]; a[e + 8] = ahi[e]; }
            v16h b = *(const v16h*)(bp + c0);
            acc = __builtin_amdgcn_wmma_f32_16x16x32_f16(
                false, a, false, b, (short)0, acc, false, false);
        }
    }

    if (xb < W) {
        _Float16* op;
        if (OUTPAD)
            op = out + ((size_t)(y + 1) * PW + (xb + 1)) * COUTSTRIDE + coutBase;
        else
            op = out + ((size_t)y * W + xb) * COUTSTRIDE + coutBase;
#pragma unroll
        for (int i = 0; i < 8; ++i) {
            float v = acc[i];
            v = v > 0.f ? v : 0.f;                // fused ReLU
            op[i + (hi ? 8 : 0)] = (_Float16)v;
        }
    }
}

// ---------------------------------------------------------------------------
// 1x1 conv 256->49 + ReLU + channel softmax. One thread per pixel.
// act: tight HWC f16 [2*NPIX][256]; kern out: f32 [2*NPIX][49]
// ---------------------------------------------------------------------------
__global__ void conv3_softmax_k(const _Float16* __restrict__ act, const float* __restrict__ w,
                                const float* __restrict__ bias, float* __restrict__ kern) {
    int p = blockIdx.x * blockDim.x + threadIdx.x;
    if (p >= 2 * NPIX) return;
    float acc[49];
#pragma unroll
    for (int t = 0; t < 49; ++t) acc[t] = bias[t];
    const _Float16* ap = act + (size_t)p * 256;
    for (int c = 0; c < 256; ++c) {
        float a = (float)ap[c];
#pragma unroll
        for (int t = 0; t < 49; ++t) acc[t] += a * w[t * 256 + c];
    }
    float m = -3.4e38f;
#pragma unroll
    for (int t = 0; t < 49; ++t) { acc[t] = acc[t] > 0.f ? acc[t] : 0.f; m = fmaxf(m, acc[t]); }
    float s = 0.f;
#pragma unroll
    for (int t = 0; t < 49; ++t) { acc[t] = __expf(acc[t] - m); s += acc[t]; }
    float inv = 1.0f / s;
    float* kp = kern + (size_t)p * 49;
#pragma unroll
    for (int t = 0; t < 49; ++t) kp[t] = acc[t] * inv;
}

// ---------------------------------------------------------------------------
// Spatially-variant 7x7: out[b,c,y,x] = sum_ij kern[b,y,x,ij]*feat[b,c,y+i-3,x+j-3]
// feat NCHW f32 (input 2), out NCHW f32 (d_out). Everything L2-resident.
// ---------------------------------------------------------------------------
__global__ void sv_conv_k(const float* __restrict__ feat, const float* __restrict__ kern,
                          float* __restrict__ out, int total) {
    int idx = blockIdx.x * blockDim.x + threadIdx.x;
    if (idx >= total) return;
    int x = idx % W;
    int y = (idx / W) % H;
    int c = (idx / NPIX) % 256;
    int b = idx / (NPIX * 256);
    const float* kp = kern + ((size_t)b * NPIX + (size_t)y * W + x) * 49;
    const float* fp = feat + (size_t)(b * 256 + c) * NPIX;
    float s = 0.f;
#pragma unroll
    for (int i = 0; i < 7; ++i) {
        int yy = y + i - 3;
        if (yy < 0 || yy >= H) continue;
#pragma unroll
        for (int j = 0; j < 7; ++j) {
            int xx = x + j - 3;
            if (xx < 0 || xx >= W) continue;
            s += kp[i * 7 + j] * fp[yy * W + xx];
        }
    }
    out[idx] = s;
}

// ---------------------------------------------------------------------------
extern "C" void kernel_launch(void* const* d_in, const int* in_sizes, int n_in,
                              void* d_out, int out_size, void* d_ws, size_t ws_size,
                              hipStream_t stream) {
    const float* cur      = (const float*)d_in[0];   // (2,128,33,33)
    const float* key      = (const float*)d_in[1];   // (2,128,33,33)
    const float* high     = (const float*)d_in[2];   // (2,256,129,129)
    const float* w_reduce = (const float*)d_in[3];   // (256,128,3,3)
    const float* b_reduce = (const float*)d_in[4];   // (256)
    const float* w_conv2  = (const float*)d_in[5];   // (256,512,3,3)
    const float* b_conv2  = (const float*)d_in[6];   // (256)
    const float* w_conv3  = (const float*)d_in[7];   // (49,256,1,1)
    const float* b_conv3  = (const float*)d_in[8];   // (49)
    float* out = (float*)d_out;                      // (2,256,129,129)

    char* ws = (char*)d_ws;
    size_t off = 0;
    auto take = [&](size_t bytes) -> char* {
        char* p = ws + off;
        off += (bytes + 255) & ~(size_t)255;
        return p;
    };
    _Float16* up_cur = (_Float16*)take((size_t)2 * PPIX * 128 * 2);   // padded HWC f16
    _Float16* up_key = (_Float16*)take((size_t)2 * PPIX * 128 * 2);   // padded HWC f16
    _Float16* concat = (_Float16*)take((size_t)2 * PPIX * 512 * 2);   // padded HWC f16 (cur|key)
    _Float16* c2out  = (_Float16*)take((size_t)2 * NPIX * 256 * 2);   // tight HWC f16
    float*    kern   = (float*)   take((size_t)2 * NPIX * 49 * 4);    // softmax kernels
    _Float16* wr16   = (_Float16*)take((size_t)256 * 1152 * 2);       // packed w_reduce
    _Float16* w216   = (_Float16*)take((size_t)256 * 4608 * 2);       // packed w_conv2

    {
        int tot = 256 * 1152;
        pack_w_k<<<(tot + 255) / 256, 256, 0, stream>>>(w_reduce, wr16, 256, 128);
    }
    {
        int tot = 256 * 4608;
        pack_w_k<<<(tot + 255) / 256, 256, 0, stream>>>(w_conv2, w216, 256, 512);
    }
    {
        int tot = 2 * PPIX * 128;   // padded coords: halo written as zeros
        upsample_k<<<(tot + 255) / 256, 256, 0, stream>>>(cur, up_cur, tot);
        upsample_k<<<(tot + 255) / 256, 256, 0, stream>>>(key, up_key, tot);
    }
    {
        int totVec = (int)(((size_t)2 * PPIX * 512 * 2) / 16);   // concat halo -> 0
        zero_fill_k<<<(totVec + 255) / 256, 256, 0, stream>>>((uint4*)concat, totVec);
    }

    // conv_reduce: cur -> concat channels [0,256), key -> [256,512)  (padded out)
    dim3 g(H * XT, 2, 2);   // pixel tiles x cout-groups(2x8 waves) x batch
    conv3x3_wmma_k<128, 512, 1><<<g, 256, 0, stream>>>(up_cur, wr16, b_reduce, concat, 0);
    conv3x3_wmma_k<128, 512, 1><<<g, 256, 0, stream>>>(up_key, wr16, b_reduce, concat, 256);
    // conv2: 512 -> 256 (tight out)
    conv3x3_wmma_k<512, 256, 0><<<g, 256, 0, stream>>>(concat, w216, b_conv2, c2out, 0);

    {
        int tot = 2 * NPIX;
        conv3_softmax_k<<<(tot + 255) / 256, 256, 0, stream>>>(c2out, w_conv3, b_conv3, kern);
    }
    {
        int tot = 2 * 256 * NPIX;
        sv_conv_k<<<(tot + 255) / 256, 256, 0, stream>>>(high, kern, out, tot);
    }
    (void)in_sizes; (void)n_in; (void)out_size; (void)ws_size;
}
